// MambaCBlock_59072980189818
// MI455X (gfx1250) — compile-verified, matrix operand order checked by reading, not run on hardware
//
#include <hip/hip_runtime.h>

// ---------------------------------------------------------------------------
// MambaCBlock for MI455X (gfx1250).
// All heavy ops (encoder convs, Mamba projections, decoder convTs) run through
// ONE f16 WMMA GEMM (fp32 accumulate). Weights are repacked once per GEMM into
// f16 [N,K] row-major so both operand tiles load as 2x global_load_b128 with
// no per-element address math. Selective scan stages the broadcast B_t/C_t
// vectors through LDS with async-to-LDS copies (ASYNCcnt pipeline).
// ---------------------------------------------------------------------------

typedef __attribute__((ext_vector_type(16))) _Float16 v16h;
typedef __attribute__((ext_vector_type(8)))  float    v8f;

#define TPB 256

// Load one 16x32 f16 tile of a row-major [R,K] matrix in WMMA operand layout:
// lanes 0-15 hold row r0+lane, K={0..7,16..23}; lanes 16-31 K={8..15,24..31}.
__device__ __forceinline__ v16h load_tile16(const _Float16* __restrict__ P,
                                            int R, int K, int r0, int k0) {
  int lane = threadIdx.x & 31;
  int hlf  = lane >> 4;
  int r = r0 + (lane & 15);
  r = (r < R) ? r : (R - 1);
  const _Float16* p = P + (size_t)r * K + k0 + 8 * hlf;
  union { uint4 q[2]; v16h v; } u;
  u.q[0] = *(const uint4*)(p);
  u.q[1] = *(const uint4*)(p + 16);
  return u.v;
}

// D[M x Nc] (cols n0.. of a [M,ldD] matrix) = A[M,K] * BT[Nc,K]^T
// Per-wave 16(M) x 128(N); 8 waves -> 32 x 512 block tile.
__global__ __launch_bounds__(TPB)
void gemm_wmma_bt16(const _Float16* __restrict__ A, const _Float16* __restrict__ BT,
                    float* __restrict__ D, int M, int N, int ldD, int K) {
  int wave = threadIdx.x >> 5;
  int lane = threadIdx.x & 31;
  int mBase = blockIdx.y * 32 + (wave & 1) * 16;
  int nBase = blockIdx.x * 512 + (wave >> 1) * 128;

  v8f acc[8] = {};
  for (int k0 = 0; k0 < K; k0 += 32) {
    if (k0 + 128 < K) {  // lowers to global_prefetch_b8
      __builtin_prefetch(A  + (size_t)(mBase + (lane & 15)) * K + k0 + 128, 0, 1);
      __builtin_prefetch(BT + (size_t)(nBase + (lane & 15)) * K + k0 + 128, 0, 1);
    }
    v16h a = load_tile16(A, M, K, mBase, k0);
#pragma unroll
    for (int j = 0; j < 8; ++j) {
      v16h b = load_tile16(BT, N, K, nBase + 16 * j, k0);
      acc[j] = __builtin_amdgcn_wmma_f32_16x16x32_f16(
          false, a, false, b, (short)0, acc[j], false, false);
    }
  }
  __builtin_amdgcn_s_wait_tensorcnt(0);  // CDNA5 wait path (no TDM in flight)

  int hlf = lane >> 4, nOff = lane & 15;
#pragma unroll
  for (int j = 0; j < 8; ++j) {
    int n = nBase + 16 * j + nOff;
    if (n >= N) continue;
#pragma unroll
    for (int v = 0; v < 8; ++v) {
      int m = mBase + v + 8 * hlf;   // C/D layout: VGPR v -> M=v (+8 lanes 16-31)
      if (m < M) D[(size_t)m * ldD + n] = acc[j][v];
    }
  }
}

// Repack weights to f16 BT[Nc, K] (rows n0..n0+Nc-1):
//   BT[nl][j] = W[(n0+nl)*sN + (j&cmask)*sIC + (j>>lc)*sK],  K = 1<<kshift
__global__ void repack_w_f16(const float* __restrict__ W, _Float16* __restrict__ BT,
                             int n0, int sN, int sIC, int sK, int lc, int cmask,
                             int kshift, int total) {
  int idx = blockIdx.x * blockDim.x + threadIdx.x;
  if (idx >= total) return;
  int nl = idx >> kshift;
  int j  = idx & ((1 << kshift) - 1);
  int n  = n0 + nl;
  BT[idx] = (_Float16)W[(size_t)n * sN + (size_t)(j & cmask) * sIC + (size_t)(j >> lc) * sK];
}

// --------------------------- elementwise kernels ---------------------------

// encoder conv1 (Cin=1, k=8, stride 4, pad 2) fused with input+1, BN, ReLU
__global__ void conv1_bn_relu(const float* __restrict__ x, const float* __restrict__ w,
                              const float* __restrict__ b, const float* __restrict__ g,
                              const float* __restrict__ be, const float* __restrict__ mean,
                              const float* __restrict__ var, float* __restrict__ out) {
  int idx = blockIdx.x * blockDim.x + threadIdx.x;   // [B=2][L=2048][C=1024]
  if (idx >= 2 * 2048 * 1024) return;
  int c  = idx & 1023;
  int t  = (idx >> 10) & 2047;
  int bb = idx >> 21;
  float acc = b[c];
  int base = 4 * t - 2;
#pragma unroll
  for (int k = 0; k < 8; ++k) {
    int s = base + k;
    if (s >= 0 && s < 8192) acc += (x[bb * 8192 + s] + 1.0f) * w[c * 8 + k];
  }
  float v = (acc - mean[c]) * rsqrtf(var[c] + 1e-5f) * g[c] + be[c];
  out[idx] = fmaxf(v, 0.0f);
}

// im2col for forward conv: act[B,Lin,C] -> patch[M, 8*C] (f16), C = 1<<lc
__global__ void im2col_conv(const float* __restrict__ act, _Float16* __restrict__ patch,
                            int Lin, int lc, int Lout, int stride, int pad, int total) {
  int idx = blockIdx.x * blockDim.x + threadIdx.x;
  if (idx >= total) return;
  int C  = 1 << lc;
  int m  = idx >> (lc + 3);
  int j  = idx & ((C << 3) - 1);
  int k  = j >> lc;
  int ic = j & (C - 1);
  int t = m % Lout, b = m / Lout;
  int s = stride * t - pad + k;
  float v = (s >= 0 && s < Lin) ? act[((size_t)(b * Lin + s) << lc) + ic] : 0.0f;
  patch[idx] = (_Float16)v;
}

// im2col for transposed conv (zero-padded K = 8*C); rows offset by m0
__global__ void im2col_convT(const float* __restrict__ act, _Float16* __restrict__ patch,
                             int m0, int Lin, int lc, int Lout, int stride, int pad,
                             int total) {
  int idx = blockIdx.x * blockDim.x + threadIdx.x;
  if (idx >= total) return;
  int C  = 1 << lc;
  int m  = idx >> (lc + 3);
  int j  = idx & ((C << 3) - 1);
  int k  = j >> lc;
  int ic = j & (C - 1);
  int mg = m0 + m;
  int t = mg % Lout, b = mg / Lout;
  int num = t + pad - k;
  float v = 0.0f;
  if (num >= 0 && (num % stride) == 0) {
    int s = num / stride;
    if (s < Lin) v = act[((size_t)(b * Lin + s) << lc) + ic];
  }
  patch[idx] = (_Float16)v;
}

// in-place bias + BN + ReLU on [M, C]
__global__ void bias_bn_relu(float* __restrict__ buf, const float* __restrict__ bias,
                             const float* __restrict__ g, const float* __restrict__ be,
                             const float* __restrict__ mean, const float* __restrict__ var,
                             int C, int total) {
  int idx = blockIdx.x * blockDim.x + threadIdx.x;
  if (idx >= total) return;
  int c = idx % C;
  float x = buf[idx] + bias[c];
  x = (x - mean[c]) * rsqrtf(var[c] + 1e-5f) * g[c] + be[c];
  buf[idx] = fmaxf(x, 0.0f);
}

// RMSNorm row-wise, fp32 in -> f16 out (feeds in_proj GEMM). One block / row.
__global__ void rmsnorm_f16(const float* __restrict__ h, const float* __restrict__ w,
                            _Float16* __restrict__ out, int C) {
  __shared__ float red[8];
  int row = blockIdx.x;
  const float* p = h + (size_t)row * C;
  float s = 0.0f;
  for (int i = threadIdx.x; i < C; i += TPB) { float v = p[i]; s += v * v; }
  for (int off = 16; off; off >>= 1) s += __shfl_down(s, off, 32);
  if ((threadIdx.x & 31) == 0) red[threadIdx.x >> 5] = s;
  __syncthreads();
  if (threadIdx.x == 0) {
    float t = 0.0f;
    for (int i = 0; i < 8; ++i) t += red[i];
    red[0] = rsqrtf(t / (float)C + 1e-5f);
  }
  __syncthreads();
  float scale = red[0];
  for (int i = threadIdx.x; i < C; i += TPB)
    out[(size_t)row * C + i] = (_Float16)(p[i] * scale * w[i]);
}

// causal depthwise conv (k=4) + SiLU on xm = xz[:, :8192]; writes fp32 + f16
__global__ void dwconv_silu(const float* __restrict__ xz, const float* __restrict__ cw,
                            const float* __restrict__ cb, float* __restrict__ xt,
                            _Float16* __restrict__ xth, int L, int total) {
  int idx = blockIdx.x * blockDim.x + threadIdx.x;   // [(b*L+t)][d], d<8192
  if (idx >= total) return;
  int d  = idx & 8191;
  int tb = idx >> 13;
  int t = tb % L, b = tb / L;
  float acc = cb[d];
#pragma unroll
  for (int k = 0; k < 4; ++k) {
    int s = t - 3 + k;
    if (s >= 0) acc += xz[((size_t)(b * L + s)) * 16384 + d] * cw[d * 4 + k];
  }
  float v = acc / (1.0f + __expf(-acc));
  xt[idx]  = v;
  xth[idx] = (_Float16)v;
}

// dt slice of x_dbl [510,288] -> f16 [510,256]
__global__ void cvt_dt(const float* __restrict__ xdbl, _Float16* __restrict__ dt16,
                       int total) {
  int idx = blockIdx.x * blockDim.x + threadIdx.x;
  if (idx >= total) return;
  int r = idx >> 8, c = idx & 255;
  dt16[idx] = (_Float16)xdbl[r * 288 + c];
}

// ------------------------------ selective scan -----------------------------
// One block = 256 channels (d) of one batch b. The per-timestep B_t/C_t
// vectors (32 floats, identical for every channel of the batch) are staged
// into LDS with async-to-LDS copies, double-buffered over 16-step windows so
// the ASYNCcnt pipeline overlaps the recurrence math.

#define SCAN_TW 16   // timesteps per LDS window

__device__ __forceinline__ void scan_stage_window(const float* __restrict__ xdbl,
                                                  float* __restrict__ dstLds,
                                                  int b, int L, int t0) {
  int tid = threadIdx.x;
  if (tid < SCAN_TW * 8) {             // 8 x b128 per timestep (32 floats)
    int ti = tid >> 3, ci = tid & 7;
    int t = t0 + ti;
    if (t < L) {
      const float* g = xdbl + ((size_t)(b * L + t)) * 288 + 256 + 4 * ci;
      unsigned int ldsOff = (unsigned int)(size_t)(dstLds + ti * 32 + 4 * ci);
      asm volatile("global_load_async_to_lds_b128 %0, %1, off"
                   :: "v"(ldsOff), "v"(g)
                   : "memory");
    }
  }
}

__global__ __launch_bounds__(TPB)
void scan_kernel(const float* __restrict__ dto, const float* __restrict__ dtp_b,
                 const float* __restrict__ a_log, const float* __restrict__ dpar,
                 const float* __restrict__ xdbl, const float* __restrict__ xt,
                 const float* __restrict__ xz, _Float16* __restrict__ yh, int L) {
  __shared__ __align__(16) float bc[2][SCAN_TW][32];  // [buf][t][B(0..15)|C(16..31)]
  int b = blockIdx.x >> 5;                       // 64 blocks: 2 batches x 32 d-chunks
  int d = ((blockIdx.x & 31) << 8) + threadIdx.x;

  float Ad[16], h[16];
#pragma unroll
  for (int n = 0; n < 16; ++n) { Ad[n] = -__expf(a_log[d * 16 + n]); h[n] = 0.0f; }
  float bias = dtp_b[d], dp = dpar[d];

  scan_stage_window(xdbl, &bc[0][0][0], b, L, 0);

  for (int t0 = 0; t0 < L; t0 += SCAN_TW) {
    int buf = (t0 / SCAN_TW) & 1;
    asm volatile("s_wait_asynccnt 0x0" ::: "memory");  // window landed in LDS
    __syncthreads();
    scan_stage_window(xdbl, &bc[buf ^ 1][0][0], b, L, t0 + SCAN_TW);  // prefetch next

    int tend = (L - t0 < SCAN_TW) ? (L - t0) : SCAN_TW;
    for (int ti = 0; ti < tend; ++ti) {
      size_t r = (size_t)(b * L + t0 + ti);
      float dt = dto[r * 8192 + d] + bias;
      float delta = (dt > 20.0f) ? dt : log1pf(__expf(dt));
      float x_t = xt[r * 8192 + d];
      float dx = delta * x_t;
      float y = 0.0f;
#pragma unroll
      for (int n = 0; n < 16; ++n) {
        float Bn = bc[buf][ti][n];
        float Cn = bc[buf][ti][16 + n];
        h[n] = __expf(delta * Ad[n]) * h[n] + dx * Bn;
        y += h[n] * Cn;
      }
      y += x_t * dp;
      float z = xz[r * 16384 + 8192 + d];
      y *= z / (1.0f + __expf(-z));
      yh[r * 8192 + d] = (_Float16)y;
    }
    __syncthreads();  // window fully consumed before its buffer is re-staged
  }
}

__global__ void add_inplace(float* __restrict__ a, const float* __restrict__ b, int total) {
  int idx = blockIdx.x * blockDim.x + threadIdx.x;
  if (idx < total) a[idx] += b[idx];
}

// decoder convT3 (OC=1, stride 4, pad 2) + final -1.0, direct
__global__ void convT3_out(const float* __restrict__ d2, const float* __restrict__ w,
                           const float* __restrict__ bias, float* __restrict__ out) {
  int idx = blockIdx.x * blockDim.x + threadIdx.x;   // [2][8192]
  if (idx >= 16384) return;
  int t = idx & 8191, b = idx >> 13;
  float acc = bias[0] - 1.0f;
  for (int k = (t + 2) & 3; k < 8; k += 4) {
    int s = (t + 2 - k) >> 2;
    if (s >= 0 && s < 2048) {
      const float* row = d2 + ((size_t)(b * 2048 + s)) * 1024;
      for (int ic = 0; ic < 1024; ++ic) acc += row[ic] * w[ic * 8 + k];
    }
  }
  out[idx] = acc;
}

// ------------------------------ host driver --------------------------------

static inline int cdiv(int a, int b) { return (a + b - 1) / b; }

// repack W (generic strides) into WB16 [Nc,K] f16, then GEMM; chunked over N.
static void run_gemm(hipStream_t stream, const _Float16* A, const float* W, float* D,
                     int M, int N, int K, int kshift,
                     int sN, int sIC, int sK, int lc, int cmask,
                     _Float16* WB16, int chunkN) {
  for (int n0 = 0; n0 < N; n0 += chunkN) {
    int Nc = (N - n0 < chunkN) ? (N - n0) : chunkN;
    int tot = Nc << kshift;
    repack_w_f16<<<cdiv(tot, TPB), TPB, 0, stream>>>(W, WB16, n0, sN, sIC, sK, lc,
                                                     cmask, kshift, tot);
    gemm_wmma_bt16<<<dim3(cdiv(Nc, 512), cdiv(M, 32)), TPB, 0, stream>>>(
        A, WB16, D + n0, M, Nc, N, K);
  }
}

extern "C" void kernel_launch(void* const* d_in, const int* in_sizes, int n_in,
                              void* d_out, int out_size, void* d_ws, size_t ws_size,
                              hipStream_t stream) {
  (void)in_sizes; (void)n_in; (void)out_size; (void)ws_size;
  const float* x     = (const float*)d_in[0];
  const float* ew1   = (const float*)d_in[1];
  const float* eb1   = (const float*)d_in[2];
  const float* eg1   = (const float*)d_in[3];
  const float* ebe1  = (const float*)d_in[4];
  const float* em1   = (const float*)d_in[5];
  const float* ev1   = (const float*)d_in[6];
  const float* ew2   = (const float*)d_in[7];
  const float* eb2   = (const float*)d_in[8];
  const float* eg2   = (const float*)d_in[9];
  const float* ebe2  = (const float*)d_in[10];
  const float* em2   = (const float*)d_in[11];
  const float* ev2   = (const float*)d_in[12];
  const float* ew3   = (const float*)d_in[13];
  const float* eb3   = (const float*)d_in[14];
  const float* eg3   = (const float*)d_in[15];
  const float* ebe3  = (const float*)d_in[16];
  const float* em3   = (const float*)d_in[17];
  const float* ev3   = (const float*)d_in[18];
  const float* normw = (const float*)d_in[19];
  const float* in_w  = (const float*)d_in[20];
  const float* cw    = (const float*)d_in[21];
  const float* cb    = (const float*)d_in[22];
  const float* xp_w  = (const float*)d_in[23];
  const float* dtp_w = (const float*)d_in[24];
  const float* dtp_b = (const float*)d_in[25];
  const float* a_log = (const float*)d_in[26];
  const float* dpar  = (const float*)d_in[27];
  const float* out_w = (const float*)d_in[28];
  const float* dw1   = (const float*)d_in[29];
  const float* db1   = (const float*)d_in[30];
  const float* dg1   = (const float*)d_in[31];
  const float* dbe1  = (const float*)d_in[32];
  const float* dm1   = (const float*)d_in[33];
  const float* dv1   = (const float*)d_in[34];
  const float* dw2   = (const float*)d_in[35];
  const float* db2   = (const float*)d_in[36];
  const float* dg2   = (const float*)d_in[37];
  const float* dbe2  = (const float*)d_in[38];
  const float* dm2   = (const float*)d_in[39];
  const float* dv2   = (const float*)d_in[40];
  const float* dw3   = (const float*)d_in[41];
  const float* db3   = (const float*)d_in[42];
  float* out = (float*)d_out;

  char* ws = (char*)d_ws;
  size_t off = 0;
  auto alloc = [&](size_t bytes) -> void* {
    void* p = ws + off;
    off += (bytes + 255) & ~(size_t)255;
    return p;
  };

  float*    E1   = (float*)   alloc((size_t)2 * 2048 * 1024 * 4);  // [2,2048,1024]
  float*    E2   = (float*)   alloc((size_t)1024 * 2048 * 4);      // [2,512,2048]
  float*    HRES = (float*)   alloc((size_t)510 * 4096 * 4);       // residual h
  _Float16* HS16 = (_Float16*)alloc((size_t)510 * 4096 * 2);
  float*    XZ   = (float*)   alloc((size_t)510 * 16384 * 4);
  float*    XT   = (float*)   alloc((size_t)510 * 8192 * 4);
  _Float16* XT16 = (_Float16*)alloc((size_t)510 * 8192 * 2);
  float*    XDBL = (float*)   alloc((size_t)510 * 288 * 4);
  _Float16* DT16 = (_Float16*)alloc((size_t)510 * 256 * 2);
  float*    DTO  = (float*)   alloc((size_t)510 * 8192 * 4);
  _Float16* Y16  = (_Float16*)alloc((size_t)510 * 8192 * 2);
  float*    PROJ = (float*)   alloc((size_t)510 * 4096 * 4);
  float*    D1   = (float*)   alloc((size_t)1024 * 2048 * 4);      // [2,512,2048]
  float*    D2   = (float*)   alloc((size_t)4096 * 1024 * 4);      // [2,2048,1024]
  _Float16* PATCH= (_Float16*)alloc((size_t)1024 * 32768 * 2);     // 64 MiB, reused
  _Float16* WB16 = (_Float16*)alloc((size_t)2048 * 16384 * 2);     // 64 MiB, reused

  dim3 blk(TPB);

  // ---- encoder ----
  conv1_bn_relu<<<cdiv(2 * 2048 * 1024, TPB), blk, 0, stream>>>(
      x, ew1, eb1, eg1, ebe1, em1, ev1, E1);

  {  // conv2: M=1024, K=8*1024=8192, N=2048
    int tot = 1024 * 8192;
    im2col_conv<<<cdiv(tot, TPB), blk, 0, stream>>>(E1, PATCH, 2048, 10, 512, 4, 2, tot);
    run_gemm(stream, PATCH, ew2, E2, 1024, 2048, 8192, 13,
             /*sN=*/1024 * 8, /*sIC=*/8, /*sK=*/1, 10, 1023, WB16, 2048);
    bias_bn_relu<<<cdiv(1024 * 2048, TPB), blk, 0, stream>>>(
        E2, eb2, eg2, ebe2, em2, ev2, 2048, 1024 * 2048);
  }
  {  // conv3: M=510, K=8*2048=16384, N=4096 -> h
    int tot = 510 * 16384;
    im2col_conv<<<cdiv(tot, TPB), blk, 0, stream>>>(E2, PATCH, 512, 11, 255, 2, 2, tot);
    run_gemm(stream, PATCH, ew3, HRES, 510, 4096, 16384, 14,
             /*sN=*/2048 * 8, /*sIC=*/8, /*sK=*/1, 11, 2047, WB16, 2048);
    bias_bn_relu<<<cdiv(510 * 4096, TPB), blk, 0, stream>>>(
        HRES, eb3, eg3, ebe3, em3, ev3, 4096, 510 * 4096);
  }

  // ---- mamba ----
  rmsnorm_f16<<<510, blk, 0, stream>>>(HRES, normw, HS16, 4096);

  // in_proj: [510,4096] x [16384,4096]^T  (projection: sN=K, sIC=1, sK=0)
  run_gemm(stream, HS16, in_w, XZ, 510, 16384, 4096, 12,
           4096, 1, 0, /*lc=*/30, /*cmask=*/4096 - 1, WB16, 8192);

  dwconv_silu<<<cdiv(510 * 8192, TPB), blk, 0, stream>>>(XZ, cw, cb, XT, XT16, 255,
                                                         510 * 8192);
  // x_proj: [510,8192] x [288,8192]^T
  run_gemm(stream, XT16, xp_w, XDBL, 510, 288, 8192, 13,
           8192, 1, 0, 30, 8192 - 1, WB16, 288);

  cvt_dt<<<cdiv(510 * 256, TPB), blk, 0, stream>>>(XDBL, DT16, 510 * 256);

  // dt_proj: [510,256] x [8192,256]^T
  run_gemm(stream, DT16, dtp_w, DTO, 510, 8192, 256, 8,
           256, 1, 0, 30, 256 - 1, WB16, 8192);

  scan_kernel<<<64, blk, 0, stream>>>(DTO, dtp_b, a_log, dpar, XDBL, XT, XZ, Y16, 255);

  // out_proj: [510,8192] x [4096,8192]^T
  run_gemm(stream, Y16, out_w, PROJ, 510, 4096, 8192, 13,
           8192, 1, 0, 30, 8192 - 1, WB16, 4096);

  add_inplace<<<cdiv(510 * 4096, TPB), blk, 0, stream>>>(PROJ, HRES, 510 * 4096);

  // ---- decoder ----
  {  // convT1: M=1024, K=8*4096=32768, N=2048
    int tot = 1024 * 32768;
    im2col_convT<<<cdiv(tot, TPB), blk, 0, stream>>>(PROJ, PATCH, 0, 255, 12, 512, 2, 2,
                                                     tot);
    run_gemm(stream, PATCH, dw1, D1, 1024, 2048, 32768, 15,
             /*sN=*/8, /*sIC=*/2048 * 8, /*sK=*/1, 12, 4095, WB16, 1024);
    bias_bn_relu<<<cdiv(1024 * 2048, TPB), blk, 0, stream>>>(
        D1, db1, dg1, dbe1, dm1, dv1, 2048, 1024 * 2048);
  }
  for (int c = 0; c < 2; ++c) {  // convT2 in 2 M-chunks: M=2048, K=16384, N=1024
    int m0 = c * 2048;
    int tot = 2048 * 16384;
    im2col_convT<<<cdiv(tot, TPB), blk, 0, stream>>>(D1, PATCH, m0, 512, 11, 2048, 4, 2,
                                                     tot);
    run_gemm(stream, PATCH, dw2, D2 + (size_t)m0 * 1024, 2048, 1024, 16384, 14,
             /*sN=*/8, /*sIC=*/1024 * 8, /*sK=*/1, 11, 2047, WB16, 1024);
  }
  bias_bn_relu<<<cdiv(4096 * 1024, TPB), blk, 0, stream>>>(
      D2, db2, dg2, dbe2, dm2, dv2, 1024, 4096 * 1024);

  convT3_out<<<cdiv(16384, TPB), blk, 0, stream>>>(D2, dw3, db3, out);
}
